// MultiDQNGNN_52012053954738
// MI455X (gfx1250) — matrix operand order repaired and checked by reading, not compiled, not run on hardware
//
#include <hip/hip_runtime.h>
#include <hip/hip_bf16.h>

typedef __attribute__((ext_vector_type(16))) _Float16 v16h;
typedef __attribute__((ext_vector_type(8)))  _Float16 v8h;
typedef __attribute__((ext_vector_type(8)))  float    v8f;
typedef __attribute__((ext_vector_type(4)))  int      v4i;

static constexpr int BATCH = 32, NAG = 64, BN = BATCH * NAG;   // 2048
static constexpr int EMBD  = 128, NACTS = 5;

// ---- CDNA5 async global->LDS path (guarded; falls back to sync copy) ----
#if defined(__AMDGCN__) && __has_builtin(__builtin_amdgcn_global_load_async_to_lds_b128) && \
    __has_builtin(__builtin_amdgcn_s_wait_asynccnt)
#define USE_ASYNC_LDS 1
#else
#define USE_ASYNC_LDS 0
#endif

static __device__ __forceinline__ void copy_b128_to_lds(const void* g, void* l) {
#if USE_ASYNC_LDS
  // builtin expects pointer-to-int4 (b128 payload): global src, LDS dst
  __builtin_amdgcn_global_load_async_to_lds_b128(
      (__attribute__((address_space(1))) v4i*)(v4i*)(void*)g,
      (__attribute__((address_space(3))) v4i*)(v4i*)l, 0, 0);
#else
  *(v8h*)l = *(const v8h*)g;
#endif
}
static __device__ __forceinline__ void lds_copy_wait() {
#if USE_ASYNC_LDS
  __builtin_amdgcn_s_wait_asynccnt(0);
#endif
}

// ---------------- weight packing ----------------
__global__ __launch_bounds__(256)
void pack_rows_k(const float* __restrict__ w, _Float16* __restrict__ o,
                 int Nrows, int Kact, int Kpad) {
  long i = (long)blockIdx.x * blockDim.x + threadIdx.x;
  long tot = (long)Nrows * Kpad;
  if (i >= tot) return;
  int n = (int)(i / Kpad), k = (int)(i % Kpad);
  o[i] = (k < Kact) ? (_Float16)w[(long)n * Kact + k] : (_Float16)0.f;
}

__global__ __launch_bounds__(256)
void pack_t_k(const float* __restrict__ w, _Float16* __restrict__ o,
              int K, int Nact, int Npad) {
  long i = (long)blockIdx.x * blockDim.x + threadIdx.x;
  long tot = (long)Npad * K;
  if (i >= tot) return;
  int n = (int)(i / K), k = (int)(i % K);
  o[i] = (n < Nact) ? (_Float16)w[(long)k * Nact + n] : (_Float16)0.f;
}

// ---------------- implicit-im2col conv as WMMA GEMM ----------------
// GEMM view: M = BN*OH*OW, K = CIN*9 (padded to KP), N = COUT.
// Weights + im2col offset table staged in LDS once per block.
template <int CIN, int COUT, int HIN, int WIN, typename TIN>
__global__ __launch_bounds__(256)
void conv_wmma_k(const TIN* __restrict__ in, const _Float16* __restrict__ wt,
                 const float* __restrict__ bias, _Float16* __restrict__ out) {
  constexpr int OH = HIN - 2, OW = WIN - 2, PIX = OH * OW;
  constexpr int KA = CIN * 9, KP = (KA + 31) & ~31, NT = COUT / 16;

  __shared__ __align__(32) _Float16 ldsW[COUT * KP];
  __shared__ int ldsOff[KP];

  const int tid = threadIdx.x;
  // stage packed weights (async DMA into LDS) + im2col offset table
  for (int c = tid; c < COUT * KP / 8; c += 256)
    copy_b128_to_lds(wt + c * 8, &ldsW[c * 8]);
  for (int k = tid; k < KP; k += 256) {
    int ci = k / 9, r = k - ci * 9, kh = r / 3, kw = r - kh * 3;
    ldsOff[k] = (ci * HIN + kh) * WIN + kw;
  }
  lds_copy_wait();
  __syncthreads();

  const int lane  = tid & 31;
  const int wave  = tid >> 5;
  const int khalf = lane >> 4;
  const int lm    = lane & 15;
  const long row0 = ((long)blockIdx.x * 8 + wave) * 16;  // grid is exact

  const long mrow = row0 + lm;
  const int  img  = (int)(mrow / PIX);
  const int  pix  = (int)(mrow % PIX);
  const int  base = (pix / OW) * WIN + (pix % OW);       // oh*WIN + ow
  const TIN* inb  = in + (long)img * CIN * HIN * WIN;

  v8f acc[NT] = {};
  for (int kb = 0; kb < KP; kb += 32) {
    v16h a;
#pragma unroll
    for (int e = 0; e < 16; ++e) {
      // A layout: 16 contiguous K at kb + khalf*16 (matches B)
      int k = kb + khalf * 16 + e;
      _Float16 v = (_Float16)0.f;
      if (k < KA) v = (_Float16)(float)inb[base + ldsOff[k]];
      a[e] = v;
    }
#pragma unroll
    for (int t = 0; t < NT; ++t) {
      // B layout: 16 contiguous K at kb + khalf*16 for column n = t*16+lm
      v16h b = *(const v16h*)&ldsW[(t * 16 + lm) * KP + kb + khalf * 16];
      acc[t] = __builtin_amdgcn_wmma_f32_16x16x32_f16(false, a, false, b,
                                                      (short)0, acc[t], false, false);
    }
  }
#pragma unroll
  for (int t = 0; t < NT; ++t) {
    int co = t * 16 + lm;
    float bv = bias[co];
#pragma unroll
    for (int r = 0; r < 8; ++r) {
      long mr = row0 + r + 8 * khalf;
      int  im = (int)(mr / PIX);
      int  pp = (int)(mr % PIX);
      float v = fmaxf(acc[t][r] + bv, 0.f);
      out[((long)im * COUT + co) * PIX + pp] = (_Float16)v;
    }
  }
}

// ---------------- generic dense WMMA GEMM ----------------
// C[M][NOUT] = act(A[M][K]f16 @ B + bias); B pre-packed as Bt[NOUT][K] f16.
// B k-tiles are double-buffered in LDS via async global->LDS DMA; all 8 waves
// in the block share the staged tile. EPI 0: f16 out (+ReLU). EPI 1: final
// f32 [M][5] with inactive-agent mask.
template <int NOUT, int NT, bool RELU, int EPI>
__global__ __launch_bounds__(256)
void gemm_wmma_k(const _Float16* __restrict__ A, const _Float16* __restrict__ Bt,
                 const float* __restrict__ bias, void* __restrict__ outp,
                 int M, int K, const unsigned char* __restrict__ inact) {
  __shared__ __align__(32) _Float16 ldsB[2][NOUT * 32];

  const int tid   = threadIdx.x;
  const int lane  = tid & 31;
  const int wave  = tid >> 5;
  const int khalf = lane >> 4;
  const int lm    = lane & 15;
  const long row0 = ((long)blockIdx.x * 8 + wave) * 16;  // grid is exact

  auto stageB = [&](int kb, int s) {
    constexpr int CH = NOUT * 4;               // 16-byte chunks per k-tile
    for (int c = tid; c < CH; c += 256) {
      int n = c >> 2, part = c & 3;
      copy_b128_to_lds(Bt + (long)n * K + kb + part * 8,
                       &ldsB[s][n * 32 + part * 8]);
    }
  };

  v8f acc[NT] = {};
  const _Float16* arow = A + (long)(row0 + lm) * K;

  stageB(0, 0);
  int cur = 0;
  for (int kb = 0; kb < K; kb += 32) {
    lds_copy_wait();       // my async chunks for buffer `cur` have landed
    __syncthreads();       // everyone's chunks landed; prev compute done
    if (kb + 32 < K) stageB(kb + 32, cur ^ 1);

    const _Float16* ap = arow + kb + khalf * 16;
    v8h lo = *(const v8h*)ap;          // K = kb + khalf*16 + 0..7
    v8h hi = *(const v8h*)(ap + 8);    // K = kb + khalf*16 + 8..15
    __builtin_prefetch(ap + 32, 0, 1);
    v16h a = __builtin_shufflevector(lo, hi, 0, 1, 2, 3, 4, 5, 6, 7,
                                             8, 9, 10, 11, 12, 13, 14, 15);
#pragma unroll
    for (int t = 0; t < NT; ++t) {
      v16h b = *(const v16h*)&ldsB[cur][(t * 16 + lm) * 32 + khalf * 16];
      acc[t] = __builtin_amdgcn_wmma_f32_16x16x32_f16(false, a, false, b,
                                                      (short)0, acc[t], false, false);
    }
    cur ^= 1;
  }

  if constexpr (EPI == 0) {
    _Float16* out = (_Float16*)outp;
#pragma unroll
    for (int t = 0; t < NT; ++t) {
      int n = t * 16 + lm;
      float bv = bias ? bias[n] : 0.f;
#pragma unroll
      for (int r = 0; r < 8; ++r) {
        long m = row0 + r + 8 * khalf;
        float v = acc[t][r] + bv;
        if (RELU) v = fmaxf(v, 0.f);
        out[m * NOUT + n] = (_Float16)v;
      }
    }
  } else {
    float* out = (float*)outp;
    int n = lm;
    if (n < NACTS) {
      float bv = bias[n];
#pragma unroll
      for (int r = 0; r < 8; ++r) {
        long m = row0 + r + 8 * khalf;      // m == b*64 + agent
        float v = acc[0][r] + bv;
        if (inact[m]) v = 0.f;
        out[m * NACTS + n] = v;
      }
    }
  }
}

// ---------------- GCN normalization + propagation ----------------
__global__ __launch_bounds__(256)
void degree_k(const float* __restrict__ adj, float* __restrict__ dis) {
  int i = blockIdx.x * blockDim.x + threadIdx.x;
  if (i >= BN) return;
  int b = i >> 6, j = i & 63;
  const float* p = adj + (long)b * NAG * NAG + j;
  float d = 0.f;
  for (int r = 0; r < NAG; ++r) d += p[r * NAG];
  dis[i] = (d > 0.f) ? rsqrtf(fmaxf(d, 1e-30f)) : 0.f;
}

// emb[b,j,d] = sum_i dis[b,i]*adj[b,i,j]*dis[b,j]*h[b,i,d] + gb[d]
__global__ __launch_bounds__(128)
void gcn_prop_k(const float* __restrict__ adj, const float* __restrict__ dis,
                const _Float16* __restrict__ h, const float* __restrict__ gb,
                _Float16* __restrict__ emb) {
  int bj = blockIdx.x;
  int b = bj >> 6, j = bj & 63, d = threadIdx.x;
  const float* ap = adj + (long)b * NAG * NAG + j;
  const float* dp = dis + b * NAG;
  const _Float16* hp = h + (long)b * NAG * EMBD + d;
  float acc = 0.f;
  for (int i = 0; i < NAG; ++i)
    acc += dp[i] * ap[i * NAG] * (float)hp[i * EMBD];
  acc = acc * dp[j] + gb[d];
  emb[(long)bj * EMBD + d] = (_Float16)acc;
}

// ---------------- launch ----------------
extern "C" void kernel_launch(void* const* d_in, const int* in_sizes, int n_in,
                              void* d_out, int out_size, void* d_ws, size_t ws_size,
                              hipStream_t stream) {
  (void)in_sizes; (void)n_in; (void)out_size; (void)ws_size;
  const float* states = (const float*)d_in[0];
  const float* adj    = (const float*)d_in[1];
  const unsigned char* inact = (const unsigned char*)d_in[2];
  const float *cw0=(const float*)d_in[3],  *cb0=(const float*)d_in[4];
  const float *cw1=(const float*)d_in[5],  *cb1=(const float*)d_in[6];
  const float *cw2=(const float*)d_in[7],  *cb2=(const float*)d_in[8];
  const float *cw3=(const float*)d_in[9],  *cb3=(const float*)d_in[10];
  const float *mw0=(const float*)d_in[11], *mb0=(const float*)d_in[12];
  const float *mw1=(const float*)d_in[13], *mb1=(const float*)d_in[14];
  const float *mw2=(const float*)d_in[15], *mb2=(const float*)d_in[16];
  const float *gw =(const float*)d_in[17], *gb =(const float*)d_in[18];
  const float *fw0=(const float*)d_in[19], *fb0=(const float*)d_in[20];
  const float *fw1=(const float*)d_in[21], *fb1=(const float*)d_in[22];
  const float *fw2=(const float*)d_in[23], *fb2=(const float*)d_in[24];

  char* ws = (char*)d_ws;
  size_t off = 0;
  auto carve = [&](size_t bytes) -> char* {
    char* p = ws + off;
    off = (off + bytes + 255) & ~(size_t)255;
    return p;
  };
  auto mx = [](size_t a, size_t b) { return a > b ? a : b; };

  // conv activation ping-pong (f16): bufA holds a0 then a2; bufB holds a1 then a3
  const size_t szA0 = (size_t)BN*16*28*28*2, szA1 = (size_t)BN*32*26*26*2;
  const size_t szA2 = (size_t)BN*16*24*24*2, szA3 = (size_t)BN*7744*2;
  _Float16* bufA = (_Float16*)carve(mx(szA0, szA2));
  _Float16* bufB = (_Float16*)carve(mx(szA1, szA3));
  _Float16* wc0  = (_Float16*)carve((size_t)16*160*2);
  _Float16* wc1  = (_Float16*)carve((size_t)32*160*2);
  _Float16* wc2  = (_Float16*)carve((size_t)16*288*2);
  _Float16* wc3  = (_Float16*)carve((size_t)16*160*2);
  _Float16* mw0T = (_Float16*)carve((size_t)128*7744*2);
  _Float16* mw1T = (_Float16*)carve((size_t)128*128*2);
  _Float16* mw2T = (_Float16*)carve((size_t)128*128*2);
  _Float16* gwT  = (_Float16*)carve((size_t)128*128*2);
  _Float16* fw0T = (_Float16*)carve((size_t)128*128*2);
  _Float16* fw1T = (_Float16*)carve((size_t)128*128*2);
  _Float16* fw2T = (_Float16*)carve((size_t)16*128*2);
  _Float16* x1    = (_Float16*)carve((size_t)BN*EMBD*2);
  _Float16* x2    = (_Float16*)carve((size_t)BN*EMBD*2);
  _Float16* feats = (_Float16*)carve((size_t)BN*EMBD*2);
  _Float16* hbuf  = (_Float16*)carve((size_t)BN*EMBD*2);
  _Float16* embb  = (_Float16*)carve((size_t)BN*EMBD*2);
  _Float16* q1    = (_Float16*)carve((size_t)BN*EMBD*2);
  _Float16* q2    = (_Float16*)carve((size_t)BN*EMBD*2);
  float*    dis   = (float*)carve((size_t)BN*4);

  auto pgrid = [](long tot) { return (int)((tot + 255) / 256); };
  // --- pack weights to f16 (convs: [Cout][K]; dense: transpose to [N][K]) ---
  pack_rows_k<<<pgrid(16*160), 256, 0, stream>>>(cw0, wc0, 16, 144, 160);
  pack_rows_k<<<pgrid(32*160), 256, 0, stream>>>(cw1, wc1, 32, 144, 160);
  pack_rows_k<<<pgrid(16*288), 256, 0, stream>>>(cw2, wc2, 16, 288, 288);
  pack_rows_k<<<pgrid(16*160), 256, 0, stream>>>(cw3, wc3, 16, 144, 160);
  pack_t_k<<<pgrid((long)128*7744), 256, 0, stream>>>(mw0, mw0T, 7744, 128, 128);
  pack_t_k<<<pgrid(128*128), 256, 0, stream>>>(mw1, mw1T, 128, 128, 128);
  pack_t_k<<<pgrid(128*128), 256, 0, stream>>>(mw2, mw2T, 128, 128, 128);
  pack_t_k<<<pgrid(128*128), 256, 0, stream>>>(gw,  gwT,  128, 128, 128);
  pack_t_k<<<pgrid(128*128), 256, 0, stream>>>(fw0, fw0T, 128, 128, 128);
  pack_t_k<<<pgrid(128*128), 256, 0, stream>>>(fw1, fw1T, 128, 128, 128);
  pack_t_k<<<pgrid(16*128),  256, 0, stream>>>(fw2, fw2T, 128, 5, 16);

  // --- conv stack (implicit-im2col WMMA); grid = M/16 waves, 8 waves/block ---
  conv_wmma_k<16, 16, 30, 30, float>
      <<<(int)((long)BN*28*28/16/8), 256, 0, stream>>>(states, wc0, cb0, bufA);
  conv_wmma_k<16, 32, 28, 28, _Float16>
      <<<(int)((long)BN*26*26/16/8), 256, 0, stream>>>(bufA, wc1, cb1, bufB);
  conv_wmma_k<32, 16, 26, 26, _Float16>
      <<<(int)((long)BN*24*24/16/8), 256, 0, stream>>>(bufB, wc2, cb2, bufA);
  conv_wmma_k<16, 16, 24, 24, _Float16>
      <<<(int)((long)BN*22*22/16/8), 256, 0, stream>>>(bufA, wc3, cb3, bufB);
  // bufB is now the flattened [2048][7744] f16 activation (NCHW == flatten order)

  // --- encoder MLP ---
  gemm_wmma_k<128, 8, true,  0><<<16, 256, 0, stream>>>(bufB, mw0T, mb0, x1,    BN, 7744, nullptr);
  gemm_wmma_k<128, 8, true,  0><<<16, 256, 0, stream>>>(x1,   mw1T, mb1, x2,    BN, 128,  nullptr);
  gemm_wmma_k<128, 8, false, 0><<<16, 256, 0, stream>>>(x2,   mw2T, mb2, feats, BN, 128,  nullptr);

  // --- GCN: h = feats @ gw ; normalized propagation + gb ---
  gemm_wmma_k<128, 8, false, 0><<<16, 256, 0, stream>>>(feats, gwT, nullptr, hbuf, BN, 128, nullptr);
  degree_k<<<pgrid(BN), 256, 0, stream>>>(adj, dis);
  gcn_prop_k<<<BN, 128, 0, stream>>>(adj, dis, hbuf, gb, embb);

  // --- DQN head + inactive mask (final f32 output [2048][5]) ---
  gemm_wmma_k<128, 8, true,  0><<<16, 256, 0, stream>>>(embb, fw0T, fb0, q1, BN, 128, nullptr);
  gemm_wmma_k<128, 8, true,  0><<<16, 256, 0, stream>>>(q1,   fw1T, fb1, q2, BN, 128, nullptr);
  gemm_wmma_k<16, 1, false, 1><<<16, 256, 0, stream>>>(q2, fw2T, fb2, d_out, BN, 128, inact);
}
